// MultiHeadAttention_81750407512394
// MI455X (gfx1250) — matrix operand order repaired and checked by reading, not compiled, not run on hardware
//
#include <hip/hip_runtime.h>
#include <hip/hip_bf16.h>

// Problem constants (match reference): B=2, S=2048, D=1024, H=16, DK=64
static constexpr int Bc = 2, Sc = 2048, Dc = 1024, Hc = 16, DKc = 64;

typedef __attribute__((ext_vector_type(16))) __bf16 v16bf;
typedef __attribute__((ext_vector_type(8)))  __bf16 v8bf;
typedef __attribute__((ext_vector_type(8)))  float  v8f;

#define WMMA_BF16(a, b, c) \
  __builtin_amdgcn_wmma_f32_16x16x32_bf16(false, (a), false, (b), (short)0, (c), false, false)

// ---- CDNA5 async global->LDS path (ASYNCcnt), gated on toolchain support ----
#if defined(__AMDGCN__) && __has_builtin(__builtin_amdgcn_global_load_async_to_lds_b128) && \
    __has_builtin(__builtin_amdgcn_s_wait_asynccnt)
#define HAS_ASYNC 1
#else
#define HAS_ASYNC 0
#endif

#if HAS_ASYNC
// Builtin signature (from compiler diagnostic): pointee is a 16-byte int vector,
// src in AS1 (__device__), dst in AS3 (__shared__), then imm offset + imm cpol.
typedef int v4i_gcc __attribute__((vector_size(16)));
__device__ __forceinline__ void async_cp16(const void* g, void* l) {
  __builtin_amdgcn_global_load_async_to_lds_b128(
      (v4i_gcc __attribute__((address_space(1)))*)(uintptr_t)g,
      (v4i_gcc __attribute__((address_space(3)))*)(unsigned int)(uintptr_t)l,
      0, 0);
}
__device__ __forceinline__ void async_wait0() { __builtin_amdgcn_s_wait_asynccnt(0); }
#endif

// Compose a v16bf A/B fragment from two 16-byte chunks.
__device__ __forceinline__ v16bf ld_v16(const __bf16* p0, const __bf16* p1) {
  v8bf a = *(const v8bf*)p0;
  v8bf b = *(const v8bf*)p1;
  v16bf r;
#pragma unroll
  for (int i = 0; i < 8; ++i) { r[i] = a[i]; r[i + 8] = b[i]; }
  return r;
}

// Reductions across the 16-lane half-wave (xor of bits 0..3 stays inside a half).
__device__ __forceinline__ float hmax16(float v) {
#pragma unroll
  for (int m = 1; m <= 8; m <<= 1) v = fmaxf(v, __shfl_xor(v, m, 32));
  return v;
}
__device__ __forceinline__ float hsum16(float v) {
#pragma unroll
  for (int m = 1; m <= 8; m <<= 1) v += __shfl_xor(v, m, 32);
  return v;
}

// ---------------------------------------------------------------------------
// Kernel 1: fp32 -> bf16 conversion (8 elements / thread, n % 8 == 0)
// ---------------------------------------------------------------------------
__global__ __launch_bounds__(256) void cvt_f32_to_bf16(const float* __restrict__ src,
                                                       __bf16* __restrict__ dst, int n) {
  int i = (blockIdx.x * blockDim.x + threadIdx.x) * 8;
  if (i + 8 <= n) {
    float4 a = *(const float4*)(src + i);
    float4 b = *(const float4*)(src + i + 4);
    v8bf o;
    o[0] = (__bf16)a.x; o[1] = (__bf16)a.y; o[2] = (__bf16)a.z; o[3] = (__bf16)a.w;
    o[4] = (__bf16)b.x; o[5] = (__bf16)b.y; o[6] = (__bf16)b.z; o[7] = (__bf16)b.w;
    *(v8bf*)(dst + i) = o;
  }
}

// ---------------------------------------------------------------------------
// Kernel 2/4: GEMM  C[M=4096, N=1024] = A[M,K=1024] * W[N,K]^T   (bf16, f32 acc)
// Block: 256 threads = 8 waves, 128x128 tile, K-step 32.
// Wave grid 4(M) x 2(N): each wave computes 32(M) x 64(N) = 2 A-frags x 4 B-frags,
// 8 WMMAs per K-step (A reused 4x, B reused 2x).
// Async path: LDS double-buffered via GLOBAL_LOAD_ASYNC_TO_LDS_B128.
// EPI=0: bf16 out scattered to [B,H,S,DK] with scale (Q/K/V projections)
// EPI=1: f32 out row-major [M, N]                     (output projection)
// ---------------------------------------------------------------------------
template <int EPI>
__global__ __launch_bounds__(256, 2) void gemm128x128(const __bf16* __restrict__ A,
                                                      const __bf16* __restrict__ W,
                                                      void* __restrict__ out, float scale) {
  constexpr int K = Dc;
  constexpr int NK = K / 32;
  constexpr int LDS_S = 40;                // 32 halves + 16B pad per row
  constexpr int NBUF = HAS_ASYNC ? 2 : 1;  // double-buffer only with async copies
  __shared__ __align__(16) __bf16 sA[NBUF][128 * LDS_S];
  __shared__ __align__(16) __bf16 sB[NBUF][128 * LDS_S];

  const int tid = threadIdx.x;
  const int wave = tid >> 5;
  const int lane = tid & 31;
  const int lid = lane & 15;
  const int sel = lane >> 4;
  const int wr = wave & 3;   // M strip: 32 rows at wr*32
  const int wc = wave >> 2;  // N strip: 64 cols at wc*64
  const int m0 = blockIdx.x * 128;
  const int n0 = blockIdx.y * 128;

  v8f acc[2][4];
#pragma unroll
  for (int u = 0; u < 2; ++u)
#pragma unroll
    for (int t = 0; t < 4; ++t)
#pragma unroll
      for (int i = 0; i < 8; ++i) acc[u][t][i] = 0.f;

  // Staging: 128 rows x 32 halves = 512 x 16B chunks per tile; 2 chunks/thread each.
  auto stage = [&](int kt, int buf) {
#pragma unroll
    for (int r = 0; r < 2; ++r) {
      int c = r * 256 + tid;
      int row = c >> 2;
      int col = (c & 3) * 8;
      const __bf16* ga = &A[(size_t)(m0 + row) * K + kt * 32 + col];
      const __bf16* gw = &W[(size_t)(n0 + row) * K + kt * 32 + col];
#if HAS_ASYNC
      async_cp16(ga, &sA[buf][row * LDS_S + col]);
      async_cp16(gw, &sB[buf][row * LDS_S + col]);
#else
      *(v8bf*)&sA[buf][row * LDS_S + col] = *(const v8bf*)ga;
      *(v8bf*)&sB[buf][row * LDS_S + col] = *(const v8bf*)gw;
#endif
    }
  };

  auto compute = [&](int buf) {
    const int ar0 = wr * 32 + lid;
    v16bf a0 = ld_v16(&sA[buf][ar0 * LDS_S + 8 * sel], &sA[buf][ar0 * LDS_S + 16 + 8 * sel]);
    v16bf a1 = ld_v16(&sA[buf][(ar0 + 16) * LDS_S + 8 * sel],
                      &sA[buf][(ar0 + 16) * LDS_S + 16 + 8 * sel]);
#pragma unroll
    for (int t = 0; t < 4; ++t) {
      const __bf16* bp = &sB[buf][(wc * 64 + t * 16 + lid) * LDS_S];
      v16bf bfm = ld_v16(bp + 8 * sel, bp + 16 + 8 * sel);
      acc[0][t] = WMMA_BF16(a0, bfm, acc[0][t]);
      acc[1][t] = WMMA_BF16(a1, bfm, acc[1][t]);
    }
  };

#if HAS_ASYNC
  stage(0, 0);
  for (int kt = 0; kt < NK; ++kt) {
    async_wait0();    // this wave's async copies for tile kt have landed
    __syncthreads();  // everyone's copies visible; everyone done reading other buf
    if (kt + 1 < NK) stage(kt + 1, (kt + 1) & 1);
    compute(kt & 1);
  }
#else
  for (int kt = 0; kt < NK; ++kt) {
    stage(kt, 0);
    __syncthreads();
    compute(0);
    __syncthreads();
  }
#endif

  // Epilogue. C/D layout: element g -> row m = g + 8*sel, col = lid.
#pragma unroll
  for (int u = 0; u < 2; ++u)
#pragma unroll
    for (int t = 0; t < 4; ++t) {
      const int n = n0 + wc * 64 + t * 16 + lid;
#pragma unroll
      for (int g = 0; g < 8; ++g) {
        const int m = m0 + wr * 32 + u * 16 + g + 8 * sel;
        const float v = acc[u][t][g] * scale;
        if (EPI == 0) {
          const int bb = m >> 11, s = m & (Sc - 1);    // S = 2048
          const int hh = n >> 6,  dk = n & (DKc - 1);  // DK = 64
          ((__bf16*)out)[((((size_t)bb * Hc + hh) * Sc + s) << 6) + dk] = (__bf16)v;
        } else {
          ((float*)out)[(size_t)m * Dc + n] = v;
        }
      }
    }
}

// ---------------------------------------------------------------------------
// Kernel 3: causal flash attention over [B,H,S,DK] bf16 Q (pre-scaled), K, V.
// Block: 256 threads = 8 waves -> 128 query rows. Key tiles of 64.
// Output O written bf16 as [B,S,H,DK] (== [B,S,D]) for the final GEMM.
// ---------------------------------------------------------------------------
__global__ __launch_bounds__(256, 1) void flash_attn(const __bf16* __restrict__ Q,
                                                     const __bf16* __restrict__ Km,
                                                     const __bf16* __restrict__ V,
                                                     __bf16* __restrict__ O) {
  constexpr int KS = 72;  // LDS row stride in halves (64 + 8 pad)
  __shared__ __align__(16) __bf16 sK[64 * KS];      // K tile  [n][dk]
  __shared__ __align__(16) __bf16 sVt[64 * KS];     // V tile transposed [dk][n]
  __shared__ __align__(16) __bf16 sP[8 * 16 * KS];  // per-wave P staging [m][n]

  const int tid = threadIdx.x;
  const int wave = tid >> 5;
  const int lane = tid & 31;
  const int lid = lane & 15;
  const int sel = lane >> 4;
  const int q0 = blockIdx.x * 128;
  const int h  = blockIdx.y;
  const int b  = blockIdx.z;
  const size_t base = ((size_t)(b * Hc + h)) * Sc * DKc;

  // Q fragments for this wave's 16 rows (Q already scaled by 1/sqrt(DK)).
  const int qrow = q0 + wave * 16 + lid;
  const __bf16* qp = Q + base + (size_t)qrow * DKc;
  v16bf qf[2];
  qf[0] = ld_v16(qp + 8 * sel,      qp + 16 + 8 * sel);
  qf[1] = ld_v16(qp + 32 + 8 * sel, qp + 48 + 8 * sel);

  v8f oacc[4];
#pragma unroll
  for (int t = 0; t < 4; ++t)
#pragma unroll
    for (int i = 0; i < 8; ++i) oacc[t][i] = 0.f;
  float mrow[8], lrow[8];
#pragma unroll
  for (int g = 0; g < 8; ++g) { mrow[g] = -3.0e38f; lrow[g] = 0.f; }

  __bf16* pw = &sP[wave * 16 * KS];
  const int jend = (q0 + 127) >> 6;  // inclusive causal bound on key tiles

  for (int j = 0; j <= jend; ++j) {
    // Stage K tile (async when available) and V tile (transposed, VGPR path).
#pragma unroll
    for (int r = 0; r < 2; ++r) {
      int c = r * 256 + tid;  // 0..511 chunks of 16B
      int row = c >> 3;
      int col = (c & 7) * 8;
      const size_t goff = base + (size_t)(j * 64 + row) * DKc + col;
#if HAS_ASYNC
      async_cp16(&Km[goff], &sK[row * KS + col]);
#else
      *(v8bf*)&sK[row * KS + col] = *(const v8bf*)&Km[goff];
#endif
      v8bf vv = *(const v8bf*)&V[goff];
#pragma unroll
      for (int i = 0; i < 8; ++i) sVt[(col + i) * KS + row] = vv[i];
      // Prefetch next tile into cache while we compute this one.
      if (j < jend) {
        __builtin_prefetch(&Km[goff + 64 * DKc], 0, 3);
        __builtin_prefetch(&V[goff + 64 * DKc], 0, 3);
      }
    }
#if HAS_ASYNC
    async_wait0();
#endif
    __syncthreads();

    // Scores: S = Q * K^T  (16 x 64 per wave, f32)
    v8f sacc[4];
#pragma unroll
    for (int t = 0; t < 4; ++t)
#pragma unroll
      for (int i = 0; i < 8; ++i) sacc[t][i] = 0.f;
#pragma unroll
    for (int kk = 0; kk < 2; ++kk) {
#pragma unroll
      for (int t = 0; t < 4; ++t) {
        const __bf16* bp = &sK[(t * 16 + lid) * KS + kk * 32];
        v16bf bfm = ld_v16(bp + 8 * sel, bp + 16 + 8 * sel);
        sacc[t] = WMMA_BF16(qf[kk], bfm, sacc[t]);
      }
    }

    // Causal mask (only tiles touching the diagonal for this wave).
    if (j * 64 + 63 > q0 + wave * 16) {
#pragma unroll
      for (int t = 0; t < 4; ++t) {
        const int kidx = j * 64 + t * 16 + lid;
#pragma unroll
        for (int g = 0; g < 8; ++g) {
          const int qidx = q0 + wave * 16 + g + 8 * sel;
          if (kidx > qidx) sacc[t][g] = -65504.f;
        }
      }
    }

    // Online softmax update per row (row's 16 cols live in one 16-lane half).
#pragma unroll
    for (int g = 0; g < 8; ++g) {
      float mx = sacc[0][g];
      mx = fmaxf(mx, sacc[1][g]); mx = fmaxf(mx, sacc[2][g]); mx = fmaxf(mx, sacc[3][g]);
      mx = hmax16(mx);
      const float mnew = fmaxf(mrow[g], mx);
      const float corr = __expf(mrow[g] - mnew);
      mrow[g] = mnew;
      float rs = 0.f;
#pragma unroll
      for (int t = 0; t < 4; ++t) {
        float p = __expf(sacc[t][g] - mnew);
        sacc[t][g] = p;
        rs += p;
      }
      rs = hsum16(rs);
      lrow[g] = lrow[g] * corr + rs;
#pragma unroll
      for (int t = 0; t < 4; ++t) oacc[t][g] *= corr;
    }

    // Re-layout P (C-layout -> A-fragment) through per-wave LDS staging.
#pragma unroll
    for (int t = 0; t < 4; ++t)
#pragma unroll
      for (int g = 0; g < 8; ++g)
        pw[(g + 8 * sel) * KS + t * 16 + lid] = (__bf16)sacc[t][g];

    // O += P * V
#pragma unroll
    for (int kk = 0; kk < 2; ++kk) {
      const __bf16* ap = &pw[lid * KS + kk * 32];
      v16bf pf = ld_v16(ap + 8 * sel, ap + 16 + 8 * sel);
#pragma unroll
      for (int t2 = 0; t2 < 4; ++t2) {
        const __bf16* vb = &sVt[(t2 * 16 + lid) * KS + kk * 32];
        v16bf vf = ld_v16(vb + 8 * sel, vb + 16 + 8 * sel);
        oacc[t2] = WMMA_BF16(pf, vf, oacc[t2]);
      }
    }
    __syncthreads();
  }

  // Normalize and store O as [B, S, H, DK] bf16.
#pragma unroll
  for (int g = 0; g < 8; ++g) {
    const float inv = 1.f / lrow[g];
    const int s = q0 + wave * 16 + g + 8 * sel;
#pragma unroll
    for (int t2 = 0; t2 < 4; ++t2) {
      const int dk = t2 * 16 + lid;
      O[((((size_t)b * Sc + s) * Hc + h) << 6) + dk] = (__bf16)(oacc[t2][g] * inv);
    }
  }
}

// ---------------------------------------------------------------------------
// Host-side orchestration
// ---------------------------------------------------------------------------
extern "C" void kernel_launch(void* const* d_in, const int* in_sizes, int n_in,
                              void* d_out, int out_size, void* d_ws, size_t ws_size,
                              hipStream_t stream) {
  (void)in_sizes; (void)n_in; (void)out_size; (void)ws_size;
  const float* x   = (const float*)d_in[0];
  const float* W_q = (const float*)d_in[1];
  const float* W_k = (const float*)d_in[2];
  const float* W_v = (const float*)d_in[3];
  const float* W_o = (const float*)d_in[4];

  const size_t nx = (size_t)Bc * Sc * Dc;  // 4,194,304
  const size_t nw = (size_t)Dc * Dc;       // 1,048,576

  char* w = (char*)d_ws;
  __bf16* xb  = (__bf16*)(w);
  __bf16* wqb = (__bf16*)(w + 2 * nx);
  __bf16* wkb = (__bf16*)(w + 2 * (nx + nw));
  __bf16* wvb = (__bf16*)(w + 2 * (nx + 2 * nw));
  __bf16* wob = (__bf16*)(w + 2 * (nx + 3 * nw));
  __bf16* Qb  = (__bf16*)(w + 2 * (nx + 4 * nw));      // [B,H,S,DK], pre-scaled
  __bf16* Kb  = (__bf16*)(w + 2 * (2 * nx + 4 * nw));
  __bf16* Vb  = (__bf16*)(w + 2 * (3 * nx + 4 * nw));
  __bf16* Ob  = (__bf16*)(w + 2 * (4 * nx + 4 * nw));  // [B,S,H,DK] == [B,S,D]

  // 1) fp32 -> bf16 conversions
  cvt_f32_to_bf16<<<(int)(nx / 8 / 256), 256, 0, stream>>>(x, xb, (int)nx);
  cvt_f32_to_bf16<<<(int)(nw / 8 / 256), 256, 0, stream>>>(W_q, wqb, (int)nw);
  cvt_f32_to_bf16<<<(int)(nw / 8 / 256), 256, 0, stream>>>(W_k, wkb, (int)nw);
  cvt_f32_to_bf16<<<(int)(nw / 8 / 256), 256, 0, stream>>>(W_v, wvb, (int)nw);
  cvt_f32_to_bf16<<<(int)(nw / 8 / 256), 256, 0, stream>>>(W_o, wob, (int)nw);

  // 2) Q/K/V projections: [4096,1024] x [1024,1024]^T, scatter to [B,H,S,DK]
  dim3 ggrid(Bc * Sc / 128, Dc / 128);  // (32, 8)
  const float qscale = 0.125f;          // 1/sqrt(DK)
  gemm128x128<0><<<ggrid, 256, 0, stream>>>(xb, wqb, (void*)Qb, qscale);
  gemm128x128<0><<<ggrid, 256, 0, stream>>>(xb, wkb, (void*)Kb, 1.0f);
  gemm128x128<0><<<ggrid, 256, 0, stream>>>(xb, wvb, (void*)Vb, 1.0f);

  // 3) Causal flash attention
  dim3 agrid(Sc / 128, Hc, Bc);  // (16, 16, 2)
  flash_attn<<<agrid, 256, 0, stream>>>(Qb, Kb, Vb, Ob);

  // 4) Output projection -> f32 d_out
  gemm128x128<1><<<ggrid, 256, 0, stream>>>(Ob, wob, d_out, 1.0f);
}
